// MultiHeadAttentionBlock_50002009260226
// MI455X (gfx1250) — matrix operand order repaired and checked by reading, not compiled, not run on hardware
//
#include <hip/hip_runtime.h>
#include <hip/hip_bf16.h>

typedef __attribute__((ext_vector_type(16))) __bf16 v16bf;
typedef __attribute__((ext_vector_type(8)))  float  v8f;

#define DMODEL 1024
#define NTOK   2048
#define BATCH  4
#define NHEAD  16
#define DHEAD  64
#define DFF    4096
#define NROWS  (BATCH*NTOK)   /* 8192 token rows */

union FragU { unsigned int u[8]; v16bf v; };

__device__ __forceinline__ unsigned short f2bf(float x){
  unsigned int u = __float_as_uint(x);
  u += 0x7FFFu + ((u >> 16) & 1u);      // round-to-nearest-even
  return (unsigned short)(u >> 16);
}

// LDS byte offset of a __shared__ object (generic LDS address: low 32 bits = LDS offset)
__device__ __forceinline__ unsigned lds_offset(const void* p){
  return (unsigned)(unsigned long long)p;
}

// CDNA5 async DMA: global -> LDS, 16 bytes per lane, tracked by ASYNCcnt
__device__ __forceinline__ void async_ld_b128(unsigned lds_off, int gbyte_off, const void* sbase){
  asm volatile("global_load_async_to_lds_b128 %0, %1, %2"
               :: "v"(lds_off), "v"(gbyte_off), "s"(sbase) : "memory");
}
__device__ __forceinline__ void wait_async_le8(){
  asm volatile("s_wait_asynccnt 0x8" ::: "memory");
}
__device__ __forceinline__ void wait_async_0(){
  asm volatile("s_wait_asynccnt 0x0" ::: "memory");
}

// A-matrix 16x32 bf16 fragment (ISA 7.12.2 16-bit A layout), K-contiguous source [row][K]
__device__ __forceinline__ v16bf frag_a(const unsigned short* base, int row, int stride, int koff){
  int lane = threadIdx.x & 31;
  int m = lane & 15, hi = lane >> 4;
  const unsigned short* p = base + (size_t)(row + m) * stride + koff;
  FragU f;
#pragma unroll
  for (int v = 0; v < 8; ++v){
    int k0 = 2*(v & 3) + 8*hi + 16*(v >> 2);
    f.u[v] = *(const unsigned int*)(p + k0);
  }
  return f.v;
}

// B-matrix 32x16 bf16 fragment, source stored transposed as [N][K] (K-contiguous)
__device__ __forceinline__ v16bf frag_b(const unsigned short* base, int ncol, int stride, int koff){
  int lane = threadIdx.x & 31;
  int n = lane & 15, hi = lane >> 4;
  const unsigned short* p = base + (size_t)(ncol + n) * stride + koff;
  FragU f;
#pragma unroll
  for (int v = 0; v < 8; ++v) f.u[v] = *(const unsigned int*)(p + 2*(v & 3) + 8*hi + 16*(v >> 2));
  return f.v;
}

__device__ __forceinline__ v8f wmma_bf16(v16bf a, v16bf b, v8f c){
  return __builtin_amdgcn_wmma_f32_16x16x32_bf16(false, a, false, b, (short)0, c, false, false);
}

// -------- weight convert + transpose: f32 [K][N] -> bf16 [N][K] --------
__global__ __launch_bounds__(256)
void k_transpose_cvt(const float* __restrict__ src, unsigned short* __restrict__ dst, int K, int N){
  int idx = blockIdx.x * 256 + threadIdx.x;
  if (idx >= K * N) return;
  int k = idx / N, n = idx - k * N;
  dst[(size_t)n * K + k] = f2bf(src[idx]);
}

// -------- layernorm (scalar alpha/beta, unbiased var ddof=1) -> bf16 --------
__global__ __launch_bounds__(256)
void k_layernorm(const float* __restrict__ x, const float* __restrict__ alpha,
                 const float* __restrict__ beta, unsigned short* __restrict__ out){
  __shared__ float red[8];
  int row = blockIdx.x;
  int tid = threadIdx.x, lane = tid & 31, wave = tid >> 5;
  const float* xr = x + (size_t)row * DMODEL;
  float v[4]; float s = 0.f;
#pragma unroll
  for (int i = 0; i < 4; ++i){ v[i] = xr[tid + i*256]; s += v[i]; }
#pragma unroll
  for (int off = 1; off < 32; off <<= 1) s += __shfl_xor(s, off, 32);
  if (lane == 0) red[wave] = s;
  __syncthreads();
  float tot = 0.f;
#pragma unroll
  for (int w = 0; w < 8; ++w) tot += red[w];
  float mu = tot * (1.0f / DMODEL);
  __syncthreads();
  float s2 = 0.f;
#pragma unroll
  for (int i = 0; i < 4; ++i){ float d = v[i] - mu; s2 += d * d; }
#pragma unroll
  for (int off = 1; off < 32; off <<= 1) s2 += __shfl_xor(s2, off, 32);
  if (lane == 0) red[wave] = s2;
  __syncthreads();
  float tot2 = 0.f;
#pragma unroll
  for (int w = 0; w < 8; ++w) tot2 += red[w];
  float rstd = rsqrtf(tot2 * (1.0f / (DMODEL - 1)) + 1e-6f);
  float a = alpha[0], bb = beta[0];
  unsigned short* orow = out + (size_t)row * DMODEL;
#pragma unroll
  for (int i = 0; i < 4; ++i) orow[tid + i*256] = f2bf(a * (v[i] - mu) * rstd + bb);
}

// -------- tiled WMMA GEMM, async-LDS double buffered: C = A(MxK) * Bt(NxK)^T --------
// Tiles: 128x128 C per block, K-step 64. 8 waves, each 64x32 of C.
// MODE 0: write bf16 scattered to (b,h,n,dh)   (QKV projections)
// MODE 1: +bias, ReLU, write bf16 row-major    (FFN1)
// MODE 2: +bias, ReLU, +resid, write f32       (FFN2 + residual)
template<int MODE>
__global__ __launch_bounds__(256)
void k_gemm(const unsigned short* __restrict__ A, const unsigned short* __restrict__ Bt,
            const float* __restrict__ bias, const float* __restrict__ resid,
            float* __restrict__ outF, unsigned short* __restrict__ outB, int N, int K){
  __shared__ unsigned short sA[2][128 * 64];
  __shared__ unsigned short sB[2][128 * 64];
  int bm = blockIdx.x, bn = blockIdx.y;
  int tid = threadIdx.x, lane = tid & 31, wave = tid >> 5;
  int wr = wave >> 2, wc = wave & 3;           // 2x4 wave grid -> 64x32 per wave
  v8f acc[4][2];
#pragma unroll
  for (int m = 0; m < 4; ++m)
#pragma unroll
    for (int n = 0; n < 2; ++n)
#pragma unroll
      for (int r = 0; r < 8; ++r) acc[m][n][r] = 0.f;

  // per-thread async-load geometry: row = tid>>1 (0..127), col chunk = (tid&1)*32 elems
  int lr = tid >> 1, lc = (tid & 1) * 32;
  const void* gAbase = (const void*)(A  + (size_t)(bm * 128) * K);
  const void* gBbase = (const void*)(Bt + (size_t)(bn * 128) * K);
  int gthr = (lr * K + lc) * 2;                       // byte offset, + kt*2 per tile
  unsigned ldsA[2] = { lds_offset(&sA[0][0]), lds_offset(&sA[1][0]) };
  unsigned ldsB[2] = { lds_offset(&sB[0][0]), lds_offset(&sB[1][0]) };
  unsigned thr_lds = (unsigned)((lr * 64 + lc) * 2);  // byte offset inside a tile

  const int T = K >> 6;
  // prologue: tile 0 -> buffer 0
  {
    unsigned la = ldsA[0] + thr_lds, lb = ldsB[0] + thr_lds;
#pragma unroll
    for (int i = 0; i < 4; ++i){
      async_ld_b128(la + 16*i, gthr + 16*i, gAbase);
      async_ld_b128(lb + 16*i, gthr + 16*i, gBbase);
    }
  }
  for (int t = 0; t < T; ++t){
    int cur = t & 1;
    if (t + 1 < T){                                   // prefetch next tile
      int go = gthr + (t + 1) * 128;                  // 64 elems * 2 bytes
      unsigned la = ldsA[cur ^ 1] + thr_lds, lb = ldsB[cur ^ 1] + thr_lds;
#pragma unroll
      for (int i = 0; i < 4; ++i){
        async_ld_b128(la + 16*i, go + 16*i, gAbase);
        async_ld_b128(lb + 16*i, go + 16*i, gBbase);
      }
      wait_async_le8();                               // current tile's 8 ops done
    } else {
      wait_async_0();
    }
    __syncthreads();                                  // tile visible to all waves
#pragma unroll
    for (int ks = 0; ks < 2; ++ks){
      v16bf af[4], bfr[2];
#pragma unroll
      for (int m = 0; m < 4; ++m) af[m]  = frag_a(&sA[cur][0], wr * 64 + m * 16, 64, ks * 32);
#pragma unroll
      for (int n = 0; n < 2; ++n) bfr[n] = frag_b(&sB[cur][0], wc * 32 + n * 16, 64, ks * 32);
#pragma unroll
      for (int m = 0; m < 4; ++m)
#pragma unroll
        for (int n = 0; n < 2; ++n)
          acc[m][n] = wmma_bf16(af[m], bfr[n], acc[m][n]);
    }
    __syncthreads();                                  // everyone done reading cur
  }

  int lm = lane & 15, hi = lane >> 4;
#pragma unroll
  for (int m = 0; m < 4; ++m)
#pragma unroll
    for (int n = 0; n < 2; ++n)
#pragma unroll
      for (int r = 0; r < 8; ++r){
        int grow = bm * 128 + wr * 64 + m * 16 + r + 8 * hi;
        int gcol = bn * 128 + wc * 32 + n * 16 + lm;
        float c = acc[m][n][r];
        if constexpr (MODE == 0){
          int b = grow >> 11, nn = grow & (NTOK - 1);
          int head = gcol >> 6, dhc = gcol & (DHEAD - 1);
          outB[(((size_t)b * NHEAD + head) * NTOK + nn) * DHEAD + dhc] = f2bf(c);
        } else if constexpr (MODE == 1){
          c += bias[gcol]; c = fmaxf(c, 0.f);
          outB[(size_t)grow * N + gcol] = f2bf(c);
        } else {
          c += bias[gcol]; c = fmaxf(c, 0.f);
          size_t gi = (size_t)grow * N + gcol;
          outF[gi] = resid[gi] + c;
        }
      }
}

// -------- flash attention, 64 queries per block, 4 waves x 16 queries --------
__global__ __launch_bounds__(128)
void k_flash(const unsigned short* __restrict__ Qb, const unsigned short* __restrict__ Kb,
             const unsigned short* __restrict__ Vb, const float* __restrict__ x,
             float* __restrict__ x1){
  __shared__ unsigned short sK[64 * 64];        // [key][dh]
  __shared__ unsigned short sVt[64 * 64];       // [dh][key]
  __shared__ unsigned short sP[4][16 * 64];     // per-wave P tile [q][key]
  int bh = blockIdx.x, qt = blockIdx.y;
  int b = bh >> 4, h = bh & 15;
  int tid = threadIdx.x, lane = tid & 31, wave = tid >> 5;
  const unsigned short* Qh = Qb + (size_t)bh * NTOK * DHEAD;
  const unsigned short* Kh = Kb + (size_t)bh * NTOK * DHEAD;
  const unsigned short* Vh = Vb + (size_t)bh * NTOK * DHEAD;
  int q0 = qt * 64 + wave * 16;

  v16bf qf0 = frag_a(Qh, q0, DHEAD, 0);
  v16bf qf1 = frag_a(Qh, q0, DHEAD, 32);

  v8f o[4];
#pragma unroll
  for (int n = 0; n < 4; ++n)
#pragma unroll
    for (int r = 0; r < 8; ++r) o[n][r] = 0.f;
  float mrow[8], lrow[8];
#pragma unroll
  for (int r = 0; r < 8; ++r){ mrow[r] = -1e30f; lrow[r] = 0.f; }
  const float scale = 0.03125f;                 // 1/sqrt(1024) — full-d scaling per reference
  int lrw = tid >> 1, lh = tid & 1;
  unsigned ldsK = lds_offset(&sK[0]);
  unsigned thrK = (unsigned)((lrw * 64 + lh * 32) * 2);
  int gthrK = (lrw * DHEAD + lh * 32) * 2;      // byte offset into K head, + kt*128 per tile

  for (int kt = 0; kt < NTOK; kt += 64){
    __syncthreads();
    // K tile via CDNA5 async LDS DMA
#pragma unroll
    for (int i = 0; i < 4; ++i)
      async_ld_b128(ldsK + thrK + 16*i, gthrK + kt*128 + 16*i, (const void*)Kh);
    // V tile: load through VGPRs and transpose-store to LDS
    {
      unsigned short vreg[32];
      const uint4* gv = (const uint4*)(Vh + (size_t)(kt + lrw) * DHEAD + lh * 32);
      *(uint4*)(vreg +  0) = gv[0];
      *(uint4*)(vreg +  8) = gv[1];
      *(uint4*)(vreg + 16) = gv[2];
      *(uint4*)(vreg + 24) = gv[3];
#pragma unroll
      for (int c = 0; c < 32; ++c) sVt[(lh * 32 + c) * 64 + lrw] = vreg[c];
    }
    wait_async_0();
    __syncthreads();

    v8f s[4];
#pragma unroll
    for (int n = 0; n < 4; ++n){
#pragma unroll
      for (int r = 0; r < 8; ++r) s[n][r] = 0.f;
      s[n] = wmma_bf16(qf0, frag_b(sK, n * 16, 64, 0),  s[n]);
      s[n] = wmma_bf16(qf1, frag_b(sK, n * 16, 64, 32), s[n]);
    }
#pragma unroll
    for (int n = 0; n < 4; ++n)
#pragma unroll
      for (int r = 0; r < 8; ++r) s[n][r] *= scale;

#pragma unroll
    for (int r = 0; r < 8; ++r){
      float vmax = fmaxf(fmaxf(s[0][r], s[1][r]), fmaxf(s[2][r], s[3][r]));
#pragma unroll
      for (int off = 1; off < 16; off <<= 1) vmax = fmaxf(vmax, __shfl_xor(vmax, off, 32));
      float mnew = fmaxf(mrow[r], vmax);
      float sc = __expf(mrow[r] - mnew);
      mrow[r] = mnew;
      float ps = 0.f;
#pragma unroll
      for (int n = 0; n < 4; ++n){ float p = __expf(s[n][r] - mnew); s[n][r] = p; ps += p; }
#pragma unroll
      for (int off = 1; off < 16; off <<= 1) ps += __shfl_xor(ps, off, 32);
      lrow[r] = lrow[r] * sc + ps;
#pragma unroll
      for (int n = 0; n < 4; ++n) o[n][r] *= sc;
    }

    {
      int lm = lane & 15, hi2 = lane >> 4;
      unsigned short* sPw = sP[wave];
#pragma unroll
      for (int n = 0; n < 4; ++n)
#pragma unroll
        for (int r = 0; r < 8; ++r)
          sPw[(r + 8 * hi2) * 64 + n * 16 + lm] = f2bf(s[n][r]);
    }
    __syncthreads();
#pragma unroll
    for (int ks = 0; ks < 2; ++ks){
      v16bf pf = frag_a(sP[wave], 0, 64, ks * 32);
#pragma unroll
      for (int n = 0; n < 4; ++n)
        o[n] = wmma_bf16(pf, frag_b(sVt, n * 16, 64, ks * 32), o[n]);
    }
  }

  int lm = lane & 15, hi2 = lane >> 4;
#pragma unroll
  for (int n = 0; n < 4; ++n)
#pragma unroll
    for (int r = 0; r < 8; ++r){
      int row = q0 + r + 8 * hi2;
      int col = n * 16 + lm;
      float val = o[n][r] / lrow[r];
      size_t gi = ((size_t)b * NTOK + row) * DMODEL + h * DHEAD + col;
      x1[gi] = x[gi] + val;
    }
}

extern "C" void kernel_launch(void* const* d_in, const int* in_sizes, int n_in,
                              void* d_out, int out_size, void* d_ws, size_t ws_size,
                              hipStream_t stream){
  (void)in_sizes; (void)n_in; (void)out_size; (void)ws_size;
  const float* x     = (const float*)d_in[0];
  const float* Wq    = (const float*)d_in[1];
  const float* Wk    = (const float*)d_in[2];
  const float* Wv    = (const float*)d_in[3];
  const float* W1    = (const float*)d_in[4];
  const float* b1    = (const float*)d_in[5];
  const float* W2    = (const float*)d_in[6];
  const float* b2    = (const float*)d_in[7];
  const float* alpha = (const float*)d_in[8];
  const float* beta  = (const float*)d_in[9];

  char* ws = (char*)d_ws;
  const size_t MB = 1024 * 1024;
  unsigned short* WQT = (unsigned short*)(ws + 0   * MB); // 2 MiB  [1024][1024]
  unsigned short* WKT = (unsigned short*)(ws + 2   * MB); // 2 MiB
  unsigned short* WVT = (unsigned short*)(ws + 4   * MB); // 2 MiB
  unsigned short* W1T = (unsigned short*)(ws + 6   * MB); // 8 MiB  [4096][1024]
  unsigned short* W2T = (unsigned short*)(ws + 14  * MB); // 8 MiB  [1024][4096]
  unsigned short* LN  = (unsigned short*)(ws + 22  * MB); // 16 MiB [8192][1024] bf16 (reused)
  unsigned short* Qb  = (unsigned short*)(ws + 38  * MB); // 16 MiB [b][h][n][dh]
  unsigned short* Kb  = (unsigned short*)(ws + 54  * MB); // 16 MiB
  unsigned short* Vb  = (unsigned short*)(ws + 70  * MB); // 16 MiB
  float*          X1  = (float*)         (ws + 86  * MB); // 32 MiB x + attn
  unsigned short* HID = (unsigned short*)(ws + 118 * MB); // 64 MiB [8192][4096]

  k_transpose_cvt<<<(1024*1024)/256, 256, 0, stream>>>(Wq, WQT, 1024, 1024);
  k_transpose_cvt<<<(1024*1024)/256, 256, 0, stream>>>(Wk, WKT, 1024, 1024);
  k_transpose_cvt<<<(1024*1024)/256, 256, 0, stream>>>(Wv, WVT, 1024, 1024);
  k_transpose_cvt<<<(1024*4096)/256, 256, 0, stream>>>(W1, W1T, 1024, 4096);
  k_transpose_cvt<<<(4096*1024)/256, 256, 0, stream>>>(W2, W2T, 4096, 1024);

  k_layernorm<<<NROWS, 256, 0, stream>>>(x, alpha, beta, LN);

  dim3 gqkv(NROWS / 128, DMODEL / 128);
  k_gemm<0><<<gqkv, 256, 0, stream>>>(LN, WQT, nullptr, nullptr, nullptr, Qb, DMODEL, DMODEL);
  k_gemm<0><<<gqkv, 256, 0, stream>>>(LN, WKT, nullptr, nullptr, nullptr, Kb, DMODEL, DMODEL);
  k_gemm<0><<<gqkv, 256, 0, stream>>>(LN, WVT, nullptr, nullptr, nullptr, Vb, DMODEL, DMODEL);

  dim3 gfa(BATCH * NHEAD, NTOK / 64);
  k_flash<<<gfa, 128, 0, stream>>>(Qb, Kb, Vb, x, X1);

  k_layernorm<<<NROWS, 256, 0, stream>>>(X1, alpha, beta, LN);

  dim3 gf1(NROWS / 128, DFF / 128);
  k_gemm<1><<<gf1, 256, 0, stream>>>(LN, W1T, b1, nullptr, nullptr, HID, DFF, DMODEL);

  dim3 gf2(NROWS / 128, DMODEL / 128);
  k_gemm<2><<<gf2, 256, 0, stream>>>(HID, W2T, b2, X1, (float*)d_out, nullptr, DMODEL, DFF);
}
